// OutliersQLinearColumn_83588653515071
// MI455X (gfx1250) — matrix operand order repaired and checked by reading, not compiled, not run on hardware
//
#include <hip/hip_runtime.h>
#include <hip/hip_bf16.h>

typedef __attribute__((ext_vector_type(16))) __bf16 v16bf;
typedef __attribute__((ext_vector_type(8)))  __bf16 v8bf;
typedef __attribute__((ext_vector_type(8)))  float  v8f;

__device__ __forceinline__ unsigned short f2bf(float f) {
    union { float f; unsigned u; } v; v.f = f;
    unsigned u = v.u;
    unsigned r = (u + 0x7FFFu + ((u >> 16) & 1u)) >> 16;   // round-to-nearest-even
    return (unsigned short)r;
}

__device__ __forceinline__ v16bf concat8(v8bf lo, v8bf hi) {
    return __builtin_shufflevector(lo, hi, 0,1,2,3,4,5,6,7,8,9,10,11,12,13,14,15);
}

// ---------------------------------------------------------------------------
// 1) Per-row binarization: q[o,k] = sign(w - mean_o) * mean|w - mean_o|
// ---------------------------------------------------------------------------
__global__ void quant_rows_kernel(const float* __restrict__ w,
                                  unsigned short* __restrict__ wq, int IC) {
    const int row = blockIdx.x;
    const int tid = threadIdx.x;
    const float* wr = w + (size_t)row * IC;
    __shared__ float red[256];

    float s = 0.f;
    for (int i = tid; i < IC; i += 256) s += wr[i];
    red[tid] = s; __syncthreads();
    for (int off = 128; off > 0; off >>= 1) {
        if (tid < off) red[tid] += red[tid + off];
        __syncthreads();
    }
    const float mean = red[0] / (float)IC;
    __syncthreads();

    float sa = 0.f;
    for (int i = tid; i < IC; i += 256) sa += fabsf(wr[i] - mean);
    red[tid] = sa; __syncthreads();
    for (int off = 128; off > 0; off >>= 1) {
        if (tid < off) red[tid] += red[tid + off];
        __syncthreads();
    }
    const float scale = red[0] / (float)IC;

    unsigned short* wqr = wq + (size_t)row * IC;
    for (int i = tid; i < IC; i += 256) {
        const float c = wr[i] - mean;
        const float q = (c > 0.f) ? scale : ((c < 0.f) ? -scale : 0.f);  // jnp.sign(0)=0
        wqr[i] = f2bf(q);
    }
}

// ---------------------------------------------------------------------------
// 2) Scatter fp32 outlier columns (in-order j loop => last duplicate wins)
// ---------------------------------------------------------------------------
__global__ void scatter_outliers_kernel(const float* __restrict__ ow,
                                        const int* __restrict__ idx,
                                        unsigned short* __restrict__ wq,
                                        int IC, int n_out) {
    const int o = blockIdx.x * blockDim.x + threadIdx.x;
    unsigned short* wqr = wq + (size_t)o * IC;
    const float* owr = ow + (size_t)o * n_out;
    for (int j = 0; j < n_out; ++j)
        wqr[idx[j]] = f2bf(owr[j]);
}

// ---------------------------------------------------------------------------
// 3) x: f32 -> bf16 (one shot, vectorized 4-wide)
// ---------------------------------------------------------------------------
__global__ void cvt_bf16_kernel(const float* __restrict__ x,
                                unsigned short* __restrict__ xb, long n4) {
    const long i = (long)blockIdx.x * blockDim.x + threadIdx.x;
    if (i >= n4) return;
    const float4 v = *(const float4*)(x + i * 4);
    ushort4 o;
    o.x = f2bf(v.x); o.y = f2bf(v.y); o.z = f2bf(v.z); o.w = f2bf(v.w);
    *(ushort4*)(xb + i * 4) = o;
}

// ---------------------------------------------------------------------------
// 4) GEMM: out = x_bf16 @ wq_bf16^T + bias
//    128x128 tile / block, 8 waves (2Mx4N), async-to-LDS double buffering.
// ---------------------------------------------------------------------------
#define BM 128
#define BN 128
#define BK 64
#define STR 72   // padded LDS row stride in halves (64 data + 8 pad)

// Issue one 128x64 bf16 tile: 4 async b128 issues per thread (per-lane 16B).
__device__ __forceinline__ void issue_async_tile(unsigned long long gbase,
                                                 unsigned lds_base,
                                                 int K, int tid) {
#pragma unroll
    for (int i = 0; i < 4; ++i) {
        const int c  = tid + i * 256;      // chunk id, 16B chunks
        const int r  = c >> 3;             // row in tile
        const int c8 = (c & 7) * 8;        // half offset within row
        const unsigned goff = (unsigned)((r * K + c8) * 2);
        const unsigned loff = lds_base + (unsigned)((r * STR + c8) * 2);
        asm volatile("global_load_async_to_lds_b128 %0, %1, %2"
                     :: "v"(loff), "v"(goff), "s"(gbase)
                     : "memory");
    }
}

__global__ void __launch_bounds__(256)
gemm_wmma_kernel(const unsigned short* __restrict__ xb,
                 const unsigned short* __restrict__ wq,
                 const float* __restrict__ bias,
                 float* __restrict__ out,
                 int M, int N, int K) {
    __shared__ unsigned short ldsA[2 * BM * STR];
    __shared__ unsigned short ldsB[2 * BN * STR];

    const int tid  = threadIdx.x;
    const int lane = tid & 31;
    const int wv   = tid >> 5;       // 0..7
    const int wm   = wv >> 2;        // 0..1  (M direction, 64 rows each)
    const int wn   = wv & 3;         // 0..3  (N direction, 32 cols each)
    const int m0   = blockIdx.y * BM;
    const int n0   = blockIdx.x * BN;
    const int hi   = lane >> 4;      // half-wave selector
    const int lr   = lane & 15;

    const unsigned ldsA0 = (unsigned)(uintptr_t)&ldsA[0];
    const unsigned ldsB0 = (unsigned)(uintptr_t)&ldsB[0];
    const unsigned long long gA = (unsigned long long)(xb + (size_t)m0 * K);
    const unsigned long long gB = (unsigned long long)(wq + (size_t)n0 * K);

    v8f acc[4][2];
#pragma unroll
    for (int mt = 0; mt < 4; ++mt)
#pragma unroll
        for (int nt = 0; nt < 2; ++nt)
            acc[mt][nt] = (v8f)(0.f);

    const int nstage = K / BK;

    // prologue: stage 0 -> buffer 0
    issue_async_tile(gA, ldsA0, K, tid);
    issue_async_tile(gB, ldsB0, K, tid);

    for (int s = 0; s < nstage; ++s) {
        const int buf = s & 1;
        const int bA  = buf * BM * STR;
        const int bB  = buf * BN * STR;

        // my async issues for this buffer are done; barrier covers other waves
        asm volatile("s_wait_asynccnt 0x0" ::: "memory");
        __syncthreads();

        // overlap: kick off next stage into the other buffer
        if (s + 1 < nstage) {
            const unsigned long long koff = (unsigned long long)((s + 1) * BK * 2);
            const int ob = (buf ^ 1);
            issue_async_tile(gA + koff, ldsA0 + (unsigned)(ob * BM * STR * 2), K, tid);
            issue_async_tile(gB + koff, ldsB0 + (unsigned)(ob * BN * STR * 2), K, tid);
        }

#pragma unroll
        for (int kk = 0; kk < BK; kk += 32) {
            // Hoist all fragments into distinct registers so the 12 ds_loads
            // can overlap and WMMAs need only partial dscnt waits.
            v16bf bfrag[2];
            v16bf afrag[4];
#pragma unroll
            for (int nt = 0; nt < 2; ++nt) {
                // B: lane<16 -> K[kk..kk+15], lane>=16 -> K[kk+16..kk+31]
                const unsigned short* p =
                    &ldsB[bB + (wn * 32 + nt * 16 + lr) * STR + kk + hi * 16];
                bfrag[nt] = concat8(*(const v8bf*)p, *(const v8bf*)(p + 8));
            }
#pragma unroll
            for (int mt = 0; mt < 4; ++mt) {
                // A: lane<16 -> K[kk..kk+15], lane>=16 -> K[kk+16..kk+31]
                const unsigned short* p =
                    &ldsA[bA + (wm * 64 + mt * 16 + lr) * STR + kk + hi * 16];
                afrag[mt] = concat8(*(const v8bf*)p, *(const v8bf*)(p + 8));
            }
#pragma unroll
            for (int mt = 0; mt < 4; ++mt)
#pragma unroll
                for (int nt = 0; nt < 2; ++nt)
                    acc[mt][nt] = __builtin_amdgcn_wmma_f32_16x16x32_bf16(
                        false, afrag[mt], false, bfrag[nt],
                        (short)0, acc[mt][nt], false, false);
        }
        // everyone done reading this buffer before stage s+2 overwrites it
        __syncthreads();
    }

    // ---- epilogue: C/D layout — lane gives N, VGPR r gives M=r(+8 hi) ----
#pragma unroll
    for (int nt = 0; nt < 2; ++nt) {
        const int n  = n0 + wn * 32 + nt * 16 + lr;
        const float bv = bias[n];
#pragma unroll
        for (int mt = 0; mt < 4; ++mt) {
            const int mb = m0 + wm * 64 + mt * 16 + hi * 8;
#pragma unroll
            for (int r = 0; r < 8; ++r)
                out[(size_t)(mb + r) * N + n] = acc[mt][nt][r] + bv;
        }
    }
}

// ---------------------------------------------------------------------------
extern "C" void kernel_launch(void* const* d_in, const int* in_sizes, int n_in,
                              void* d_out, int out_size, void* d_ws, size_t ws_size,
                              hipStream_t stream) {
    const float* x    = (const float*)d_in[0];
    const float* w    = (const float*)d_in[1];
    const float* bias = (const float*)d_in[2];
    const float* ow   = (const float*)d_in[3];
    const int*   idx  = (const int*)d_in[4];
    float* out        = (float*)d_out;

    const int OC    = in_sizes[2];                   // 4096
    const int IC    = in_sizes[1] / OC;              // 4096
    const int M     = (int)((long)in_sizes[0] / IC); // 8192 (= B*S)
    const int n_out = in_sizes[4];                   // 204
    const int N     = OC;
    const int K     = IC;

    unsigned short* xb = (unsigned short*)d_ws;      // M*K bf16 (64 MB)
    unsigned short* wq = xb + (size_t)M * K;         // OC*IC bf16 (32 MB)

    quant_rows_kernel<<<OC, 256, 0, stream>>>(w, wq, IC);
    scatter_outliers_kernel<<<OC / 256, 256, 0, stream>>>(ow, idx, wq, IC, n_out);
    const long n4 = (long)M * K / 4;
    cvt_bf16_kernel<<<(unsigned)((n4 + 255) / 256), 256, 0, stream>>>(x, xb, n4);
    dim3 grid(N / BN, M / BM);
    gemm_wmma_kernel<<<grid, 256, 0, stream>>>(xb, wq, bias, out, M, N, K);
}
